// A2M_5738076307533
// MI455X (gfx1250) — matrix operand order repaired and checked by reading, not compiled, not run on hardware
//
#include <hip/hip_runtime.h>
#include <hip/hip_bf16.h>

// ---------------------------------------------------------------------------
// Types for CDNA5 WMMA (wave32, 16x16x32 f16 -> f32)
// ---------------------------------------------------------------------------
typedef _Float16 h4  __attribute__((ext_vector_type(4)));
typedef _Float16 h8  __attribute__((ext_vector_type(8)));
typedef _Float16 v16h __attribute__((ext_vector_type(16)));
typedef float    v8f __attribute__((ext_vector_type(8)));
typedef float    f4  __attribute__((ext_vector_type(4)));

#define N_MAP 131072
#define N_ACT 1024
#define D     128
#define E_TOT 262144
#define EPSG  1e-5f

// ---------------------------------------------------------------------------
// Fragment builders (ISA 7.12.2 layouts)
// A 16x32 f16: lane l -> row l&15; lanes<16 hold K {0..7,16..23},
// lanes>=16 hold K {8..15,24..31}  -> two 16B LDS loads.
// B 32x16 f16: lane l -> col l&15; lanes<16 hold K {0..7,16..23},
// lanes>=16 hold K {8..15,24..31} -> two 16B global loads from row-major W[n][k].
// ---------------------------------------------------------------------------
__device__ __forceinline__ v16h frag16(const _Float16* p0, const _Float16* p1) {
  h8 a = *(const h8*)p0;
  h8 b = *(const h8*)p1;
  v16h r;
#pragma unroll
  for (int i = 0; i < 8; ++i) { r[i] = a[i]; r[i + 8] = b[i]; }
  return r;
}

__device__ __forceinline__ v16h a_frag(const _Float16* tile, int lda, int k0) {
  int l = threadIdx.x & 31;
  const _Float16* p = tile + (l & 15) * lda + k0 + ((l >> 4) << 3);
  return frag16(p, p + 16);
}

__device__ __forceinline__ v16h b_frag(const _Float16* W, int KW, int n0, int k0) {
  int l = threadIdx.x & 31;
  const _Float16* p = W + (n0 + (l & 15)) * KW + k0 + ((l >> 4) << 3);
  return frag16(p, p + 16);
}

// Accumulate a 16xKA tile (A in LDS, lda halves) against W[n][kw_off+k]
// (f16 global, row length KW) into 8 column-tile accumulators.
template <int KA, int LDA>
__device__ __forceinline__ void gemm_acc(const _Float16* tile, const _Float16* W,
                                         int kw_off, int KW, v8f acc[8]) {
#pragma unroll
  for (int k0 = 0; k0 < KA; k0 += 32) {
    v16h a = a_frag(tile, LDA, k0);
#pragma unroll
    for (int ct = 0; ct < 8; ++ct) {
      v16h b = b_frag(W, KW, ct * 16, kw_off + k0);
      acc[ct] = __builtin_amdgcn_wmma_f32_16x16x32_f16(
          false, a, false, b, (short)0, acc[ct], false, false);
    }
  }
}

// GroupNorm(1,128) + affine (+optional ReLU) fully in-wave.
// C layout: VGPR j / lane l -> row j+8*(l>>4), col (l&15)+16*ct.
// For fixed j all 16 lanes of a half hold the SAME row -> shfl-xor reduce.
__device__ __forceinline__ void gn_rows(v8f acc[8], const float* g, const float* b,
                                        bool relu) {
  int l = threadIdx.x & 31;
  int n0 = l & 15;
  float gs[8], bs[8];
#pragma unroll
  for (int ct = 0; ct < 8; ++ct) { gs[ct] = g[n0 + ct * 16]; bs[ct] = b[n0 + ct * 16]; }
#pragma unroll
  for (int j = 0; j < 8; ++j) {
    float s = 0.f, q = 0.f;
#pragma unroll
    for (int ct = 0; ct < 8; ++ct) { float v = acc[ct][j]; s += v; q += v * v; }
#pragma unroll
    for (int m = 1; m < 16; m <<= 1) {
      s += __shfl_xor(s, m, 32);
      q += __shfl_xor(q, m, 32);
    }
    float mean = s * 0.0078125f;
    float inv = rsqrtf(q * 0.0078125f - mean * mean + EPSG);
#pragma unroll
    for (int ct = 0; ct < 8; ++ct) {
      float v = (acc[ct][j] - mean) * inv * gs[ct] + bs[ct];
      acc[ct][j] = relu ? fmaxf(v, 0.f) : v;
    }
  }
}

__device__ __forceinline__ void store_lds_h(_Float16* tile, const v8f acc[8]) {
  int l = threadIdx.x & 31;
  int rb = (l >> 4) * 8, n0 = l & 15;
#pragma unroll
  for (int j = 0; j < 8; ++j)
#pragma unroll
    for (int ct = 0; ct < 8; ++ct)
      tile[(rb + j) * 136 + n0 + ct * 16] = (_Float16)acc[ct][j];
}

// f32 global rows -> f16 LDS tile (per-wave: 2 lanes per row, 64 cols each)
__device__ __forceinline__ void fill_tile128(const float* src_base, _Float16* tile) {
  int l = threadIdx.x & 31;
  int m = l & 15, part = l >> 4;
  const float* src = src_base + m * 128 + part * 64;
  _Float16* dst = tile + m * 136 + part * 64;
#pragma unroll
  for (int c = 0; c < 16; ++c) {
    f4 v = *(const f4*)(src + c * 4);
    h4 h;
    h[0] = (_Float16)v[0]; h[1] = (_Float16)v[1];
    h[2] = (_Float16)v[2]; h[3] = (_Float16)v[3];
    *(h4*)(dst + c * 4) = h;
  }
}

// ---------------------------------------------------------------------------
// Converters
// ---------------------------------------------------------------------------
__global__ void cvt16_kernel(const float* __restrict__ s, _Float16* __restrict__ d, int n) {
  int i = blockIdx.x * 256 + threadIdx.x;
  if (i < n) d[i] = (_Float16)s[i];
}

// meta_W [128,132] -> padded f16 [128,160] (zeros beyond 132)
__global__ void cvt_metaW_kernel(const float* __restrict__ s, _Float16* __restrict__ d) {
  int i = blockIdx.x * 256 + threadIdx.x;
  if (i >= 128 * 160) return;
  int r = i / 160, k = i - r * 160;
  d[i] = (k < 132) ? (_Float16)s[r * 132 + k] : (_Float16)0.f;
}

// ---------------------------------------------------------------------------
// meta fuse: feat = relu(GN(concat(feat,turn,ctrl,its) @ metaW.T))
// 256 thr = 8 waves, each wave one 16-row tile; K padded to 160.
// ---------------------------------------------------------------------------
__global__ void __launch_bounds__(256)
meta_kernel(const float* __restrict__ feat, const float* __restrict__ turn,
            const float* __restrict__ control, const float* __restrict__ intersect,
            const _Float16* __restrict__ Wh, const float* __restrict__ g,
            const float* __restrict__ b, float* __restrict__ out) {
  __shared__ _Float16 tiles[8][16 * 168];
  int wid = threadIdx.x >> 5, l = threadIdx.x & 31;
  int row0 = (blockIdx.x * 8 + wid) * 16;
  _Float16* tile = tiles[wid];
  int m = l & 15, part = l >> 4;
  int r = row0 + m;
  const float* src = feat + (long)r * 128;
  _Float16* dst = tile + m * 168;
  if (part == 0) {
#pragma unroll
    for (int c = 0; c < 20; ++c) {
      f4 v = *(const f4*)(src + c * 4);
      h4 h; h[0] = (_Float16)v[0]; h[1] = (_Float16)v[1];
      h[2] = (_Float16)v[2]; h[3] = (_Float16)v[3];
      *(h4*)(dst + c * 4) = h;
    }
  } else {
#pragma unroll
    for (int c = 20; c < 32; ++c) {
      f4 v = *(const f4*)(src + c * 4);
      h4 h; h[0] = (_Float16)v[0]; h[1] = (_Float16)v[1];
      h[2] = (_Float16)v[2]; h[3] = (_Float16)v[3];
      *(h4*)(dst + c * 4) = h;
    }
    h4 e;
    e[0] = (_Float16)turn[r * 2];
    e[1] = (_Float16)turn[r * 2 + 1];
    e[2] = (_Float16)control[r];
    e[3] = (_Float16)intersect[r];
    *(h4*)(dst + 128) = e;
    h4 z = (h4)(_Float16)0.f;
#pragma unroll
    for (int c = 33; c < 40; ++c) *(h4*)(dst + c * 4) = z;
  }
  __builtin_amdgcn_wave_barrier();
  v8f acc[8] = {};
  gemm_acc<160, 168>(tile, Wh, 0, 160, acc);
  gn_rows(acc, g, b, true);
  int rb = part * 8, n0 = m;
#pragma unroll
  for (int j = 0; j < 8; ++j) {
    long rr = row0 + rb + j;
#pragma unroll
    for (int ct = 0; ct < 8; ++ct) out[rr * 128 + n0 + ct * 16] = acc[ct][j];
  }
}

// ---------------------------------------------------------------------------
// self path: out = in @ W.T   (128x128, no norm)
// ---------------------------------------------------------------------------
__global__ void __launch_bounds__(256)
self_gemm_kernel(const float* __restrict__ in, const _Float16* __restrict__ Wh,
                 float* __restrict__ out) {
  __shared__ _Float16 tiles[8][16 * 136];
  int wid = threadIdx.x >> 5, l = threadIdx.x & 31;
  int row0 = (blockIdx.x * 8 + wid) * 16;
  _Float16* tile = tiles[wid];
  fill_tile128(in + (long)row0 * 128, tile);
  __builtin_amdgcn_wave_barrier();
  v8f acc[8] = {};
  gemm_acc<128, 136>(tile, Wh, 0, 128, acc);
  int rb = (l >> 4) * 8, n0 = l & 15;
#pragma unroll
  for (int j = 0; j < 8; ++j) {
    long rr = row0 + rb + j;
#pragma unroll
    for (int ct = 0; ct < 8; ++ct) out[rr * 128 + n0 + ct * 16] = acc[ct][j];
  }
}

// ---------------------------------------------------------------------------
// Fully fused edge pipeline (one wave = 16 edges). 64 thr = 2 waves/block.
// actors[wi] gather uses the CDNA5 async memory->LDS path (ASYNCcnt),
// issued right after indices are known and overlapped with two GEMMs.
// ---------------------------------------------------------------------------
__global__ void __launch_bounds__(64)
edge_kernel(const float* __restrict__ feat, const float* __restrict__ map_ctrs,
            const float* __restrict__ actor_ctrs, const _Float16* __restrict__ actors_h,
            const int* __restrict__ hi, const int* __restrict__ wi,
            const float* __restrict__ dW1, const float* __restrict__ db1,
            const _Float16* __restrict__ dW2h, const float* __restrict__ dg2,
            const float* __restrict__ dbt2, const _Float16* __restrict__ qWh,
            const float* __restrict__ qg, const float* __restrict__ qb,
            const _Float16* __restrict__ cW1h, const float* __restrict__ cg1,
            const float* __restrict__ cb1, const _Float16* __restrict__ cW2h,
            float* __restrict__ a_buf) {
  __shared__ _Float16 T[2][4][16 * 136];
  __shared__ int HIS[2][16];
  __shared__ int WIS[2][16];
  __shared__ float D0[2][16][2];
  int wid = threadIdx.x >> 5, l = threadIdx.x & 31;
  int e0 = (blockIdx.x * 2 + wid) * 16;
  _Float16* T0 = T[wid][0];
  _Float16* T1 = T[wid][1];
  _Float16* T2 = T[wid][2];
  _Float16* T3 = T[wid][3];

  if (l < 16) {
    int e = e0 + l;
    int hv = hi[e], wv = wi[e];
    HIS[wid][l] = hv; WIS[wid][l] = wv;
    D0[wid][l][0] = map_ctrs[hv * 2] - actor_ctrs[wv * 2];
    D0[wid][l][1] = map_ctrs[hv * 2 + 1] - actor_ctrs[wv * 2 + 1];
  }
  __builtin_amdgcn_wave_barrier();

  // Kick off async actors_h[wi] -> T3 (f16, no conversion): overlaps with the
  // d2 and q GEMMs below; consumed only at the third cW1 segment.
  {
    int m = l & 15, part = l >> 4;
    const _Float16* src = actors_h + (long)WIS[wid][m] * 128 + part * 64;
    _Float16* dst = T3 + m * 136 + part * 64;
#pragma unroll
    for (int c = 0; c < 8; ++c) {
      unsigned ldsaddr = (unsigned)(uintptr_t)(dst + c * 8);
      const void* gaddr = (const void*)(src + c * 8);
      asm volatile("global_load_async_to_lds_b128 %0, %1, off"
                   :: "v"(ldsaddr), "v"(gaddr) : "memory");
    }
  }

  // d1 = relu(d0 @ dW1.T + db1): K=2, VALU. Lane handles 4 output cols.
  {
    int n = l * 4;
    f4 w0, w1, bb;
#pragma unroll
    for (int t = 0; t < 4; ++t) {
      w0[t] = dW1[(n + t) * 2];
      w1[t] = dW1[(n + t) * 2 + 1];
      bb[t] = db1[n + t];
    }
#pragma unroll
    for (int m = 0; m < 16; ++m) {
      float x0 = D0[wid][m][0], x1 = D0[wid][m][1];
      h4 hv;
#pragma unroll
      for (int t = 0; t < 4; ++t)
        hv[t] = (_Float16)fmaxf(fmaf(x0, w0[t], fmaf(x1, w1[t], bb[t])), 0.f);
      *(h4*)(T0 + m * 136 + n) = hv;
    }
  }
  __builtin_amdgcn_wave_barrier();

  // d2 = relu(GN(d1 @ dW2.T)) -> T1
  {
    v8f acc[8] = {};
    gemm_acc<128, 136>(T0, dW2h, 0, 128, acc);
    gn_rows(acc, dg2, dbt2, true);
    store_lds_h(T1, acc);
  }
  __builtin_amdgcn_wave_barrier();

  // gather feat[hi] (f32 -> f16) into T0
  {
    int m = l & 15, part = l >> 4;
    const float* src = feat + (long)HIS[wid][m] * 128 + part * 64;
    _Float16* dst = T0 + m * 136 + part * 64;
#pragma unroll
    for (int c = 0; c < 16; ++c) {
      f4 v = *(const f4*)(src + c * 4);
      h4 h; h[0] = (_Float16)v[0]; h[1] = (_Float16)v[1];
      h[2] = (_Float16)v[2]; h[3] = (_Float16)v[3];
      *(h4*)(dst + c * 4) = h;
    }
  }
  __builtin_amdgcn_wave_barrier();

  // q = relu(GN(feat[hi] @ qW.T)) -> T2
  {
    v8f acc[8] = {};
    gemm_acc<128, 136>(T0, qWh, 0, 128, acc);
    gn_rows(acc, qg, qb, true);
    store_lds_h(T2, acc);
  }
  __builtin_amdgcn_wave_barrier();

  // drain the async actors gather before its tile is consumed
  asm volatile("s_wait_asynccnt 0x0" ::: "memory");
  __builtin_amdgcn_wave_barrier();

  // c = relu(GN(concat(d2,q,act) @ cW1.T)): K=384 as three 128-K segments
  {
    v8f acc[8] = {};
    gemm_acc<128, 136>(T1, cW1h, 0, 384, acc);
    gemm_acc<128, 136>(T2, cW1h, 128, 384, acc);
    gemm_acc<128, 136>(T3, cW1h, 256, 384, acc);
    gn_rows(acc, cg1, cb1, true);
    store_lds_h(T1, acc);
  }
  __builtin_amdgcn_wave_barrier();

  // c2 = c @ cW2.T ; scatter-add into a_buf[hi]
  {
    v8f acc[8] = {};
    gemm_acc<128, 136>(T1, cW2h, 0, 128, acc);
    int rb = (l >> 4) * 8, n0 = l & 15;
#pragma unroll
    for (int j = 0; j < 8; ++j) {
      long row = HIS[wid][rb + j];
      float* base = a_buf + row * 128 + n0;
#pragma unroll
      for (int ct = 0; ct < 8; ++ct) unsafeAtomicAdd(base + ct * 16, acc[ct][j]);
    }
  }
}

// ---------------------------------------------------------------------------
// a = relu(GN(a)*g+b) in place; one wave per row (4 f32/lane).
// ---------------------------------------------------------------------------
__global__ void __launch_bounds__(256)
rownorm_kernel(float* __restrict__ a, const float* __restrict__ g,
               const float* __restrict__ b) {
  int wid = threadIdx.x >> 5, l = threadIdx.x & 31;
  long row = blockIdx.x * 8 + wid;
  float* p = a + row * 128 + l * 4;
  f4 v = *(const f4*)p;
  float s = v[0] + v[1] + v[2] + v[3];
  float q = v[0] * v[0] + v[1] * v[1] + v[2] * v[2] + v[3] * v[3];
#pragma unroll
  for (int m = 1; m < 32; m <<= 1) {
    s += __shfl_xor(s, m, 32);
    q += __shfl_xor(q, m, 32);
  }
  float mean = s * 0.0078125f;
  float inv = rsqrtf(q * 0.0078125f - mean * mean + EPSG);
#pragma unroll
  for (int k = 0; k < 4; ++k) {
    float x = (v[k] - mean) * inv * g[l * 4 + k] + b[l * 4 + k];
    v[k] = fmaxf(x, 0.f);
  }
  *(f4*)p = v;
}

// ---------------------------------------------------------------------------
// feat = relu(GN(h @ lW.T)*lg+lb + feat)   (in-place on feat)
// ---------------------------------------------------------------------------
__global__ void __launch_bounds__(256)
final_kernel(const float* __restrict__ h, const _Float16* __restrict__ Wh,
             const float* __restrict__ g, const float* __restrict__ b,
             float* __restrict__ feat) {
  __shared__ _Float16 tiles[8][16 * 136];
  int wid = threadIdx.x >> 5, l = threadIdx.x & 31;
  int row0 = (blockIdx.x * 8 + wid) * 16;
  _Float16* tile = tiles[wid];
  fill_tile128(h + (long)row0 * 128, tile);
  __builtin_amdgcn_wave_barrier();
  v8f acc[8] = {};
  gemm_acc<128, 136>(tile, Wh, 0, 128, acc);
  gn_rows(acc, g, b, false);
  int rb = (l >> 4) * 8, n0 = l & 15;
#pragma unroll
  for (int j = 0; j < 8; ++j) {
    long rr = row0 + rb + j;
#pragma unroll
    for (int ct = 0; ct < 8; ++ct) {
      long idx = rr * 128 + n0 + ct * 16;
      feat[idx] = fmaxf(acc[ct][j] + feat[idx], 0.f);
    }
  }
}

// ---------------------------------------------------------------------------
// Host-side launch
// ---------------------------------------------------------------------------
extern "C" void kernel_launch(void* const* d_in, const int* in_sizes, int n_in,
                              void* d_out, int out_size, void* d_ws, size_t ws_size,
                              hipStream_t stream) {
  (void)in_sizes; (void)n_in; (void)out_size; (void)ws_size;
  const float* feat_in   = (const float*)d_in[0];
  const float* turn      = (const float*)d_in[1];
  const float* control   = (const float*)d_in[2];
  const float* intersect = (const float*)d_in[3];
  const float* map_ctrs  = (const float*)d_in[4];
  const float* actors    = (const float*)d_in[5];
  const float* actor_ctrs= (const float*)d_in[6];
  const int*   hi        = (const int*)d_in[7];
  const int*   wi        = (const int*)d_in[8];
  const float* meta_W    = (const float*)d_in[9];
  const float* meta_g    = (const float*)d_in[10];
  const float* meta_b    = (const float*)d_in[11];
  const float* dW1       = (const float*)d_in[12];
  const float* db1       = (const float*)d_in[13];
  const float* dW2       = (const float*)d_in[14];
  const float* dg2       = (const float*)d_in[15];
  const float* dbt2      = (const float*)d_in[16];
  const float* qW        = (const float*)d_in[17];
  const float* qg        = (const float*)d_in[18];
  const float* qb        = (const float*)d_in[19];
  const float* cW1       = (const float*)d_in[20];
  const float* cg1       = (const float*)d_in[21];
  const float* cb1       = (const float*)d_in[22];
  const float* cW2       = (const float*)d_in[23];
  const float* aW        = (const float*)d_in[24];
  const float* ng_g      = (const float*)d_in[25];
  const float* ng_b      = (const float*)d_in[26];
  const float* lW        = (const float*)d_in[27];
  const float* lg        = (const float*)d_in[28];
  const float* lb        = (const float*)d_in[29];
  float* out = (float*)d_out;

  // workspace layout
  char* ws = (char*)d_ws;
  float*     a_buf    = (float*)ws;                                   // 64 MB
  _Float16*  actors_h = (_Float16*)(ws + (size_t)N_MAP * 128 * 4);    // 256 KB
  _Float16*  metaWh   = actors_h + N_ACT * 128;                       // 128*160
  _Float16*  wbase    = metaWh + 128 * 160;
  // per block i (131072 halves): dW2h qWh cW1h cW2h aWh lWh
  const int OFF_DW2 = 0, OFF_QW = 16384, OFF_CW1 = 32768,
            OFF_CW2 = 81920, OFF_AW = 98304, OFF_LW = 114688;

  // --- weight / actor conversions ---
  cvt16_kernel<<<(N_ACT * 128 + 255) / 256, 256, 0, stream>>>(actors, actors_h, N_ACT * 128);
  cvt_metaW_kernel<<<(128 * 160 + 255) / 256, 256, 0, stream>>>(meta_W, metaWh);
  for (int i = 0; i < 2; ++i) {
    _Float16* Wi = wbase + (size_t)i * 131072;
    cvt16_kernel<<<64, 256, 0, stream>>>(dW2 + i * 16384, Wi + OFF_DW2, 16384);
    cvt16_kernel<<<64, 256, 0, stream>>>(qW  + i * 16384, Wi + OFF_QW, 16384);
    cvt16_kernel<<<192, 256, 0, stream>>>(cW1 + i * 49152, Wi + OFF_CW1, 49152);
    cvt16_kernel<<<64, 256, 0, stream>>>(cW2 + i * 16384, Wi + OFF_CW2, 16384);
    cvt16_kernel<<<64, 256, 0, stream>>>(aW  + i * 16384, Wi + OFF_AW, 16384);
    cvt16_kernel<<<64, 256, 0, stream>>>(lW  + i * 16384, Wi + OFF_LW, 16384);
  }

  // --- meta fuse: feat -> out ---
  meta_kernel<<<N_MAP / 128, 256, 0, stream>>>(feat_in, turn, control, intersect,
                                               metaWh, meta_g, meta_b, out);

  // --- two attention blocks ---
  for (int i = 0; i < 2; ++i) {
    _Float16* Wi = wbase + (size_t)i * 131072;
    // a = feat @ aW.T
    self_gemm_kernel<<<N_MAP / 128, 256, 0, stream>>>(out, Wi + OFF_AW, a_buf);
    // fused edge pipeline, scatter-add into a
    edge_kernel<<<E_TOT / 32, 64, 0, stream>>>(
        out, map_ctrs, actor_ctrs, actors_h, hi, wi,
        dW1 + i * 256, db1 + i * 128,
        Wi + OFF_DW2, dg2 + i * 128, dbt2 + i * 128,
        Wi + OFF_QW, qg + i * 128, qb + i * 128,
        Wi + OFF_CW1, cg1 + i * 128, cb1 + i * 128,
        Wi + OFF_CW2, a_buf);
    // a = relu(gn(a))
    rownorm_kernel<<<N_MAP / 8, 256, 0, stream>>>(a_buf, ng_g + i * 128, ng_b + i * 128);
    // feat = relu(gn(a @ lW.T) + feat)
    final_kernel<<<N_MAP / 128, 256, 0, stream>>>(a_buf, Wi + OFF_LW,
                                                  lg + i * 128, lb + i * 128, out);
  }
}
